// DiffeomorphicLearnerTorch_11613591568470
// MI455X (gfx1250) — matrix operand order repaired and checked by reading, not compiled, not run on hardware
//
#include <hip/hip_runtime.h>

#define NPTS 4096
#define DDIM 256
#define TSTEPS 8

typedef __attribute__((ext_vector_type(16))) __bf16 v16bf;
typedef __attribute__((ext_vector_type(8)))  __bf16 v8bf;
typedef __attribute__((ext_vector_type(8)))  float  v8f;

union BF16x16 { v16bf v; v8bf h[2]; };

__device__ __forceinline__ unsigned short f32_to_bf16_rne(float f) {
    unsigned int u = __float_as_uint(f);
    u += 0x7FFFu + ((u >> 16) & 1u);
    return (unsigned short)(u >> 16);
}

// ---- per-step prep: bf16 copy of Z + row squared norms (fp32, exact) ----
__global__ void prepz_kernel(const float* __restrict__ Z,
                             unsigned short* __restrict__ Zbf,
                             float* __restrict__ sq) {
    const int wave = threadIdx.x >> 5;
    const int lane = threadIdx.x & 31;
    const int row  = blockIdx.x * 8 + wave;
    const float* zr = Z + (size_t)row * DDIM;
    unsigned short* zb = Zbf + (size_t)row * DDIM;
    float acc = 0.f;
#pragma unroll
    for (int c = 0; c < DDIM / 32; ++c) {
        float v = zr[lane + c * 32];
        acc += v * v;
        zb[lane + c * 32] = f32_to_bf16_rne(v);
    }
#pragma unroll
    for (int off = 16; off > 0; off >>= 1)
        acc += __shfl_xor(acc, off, 32);
    if (lane == 0) sq[row] = acc;
}

// ---- one-time: AbfT[t][d][n] = bf16(A[t][n][d]) (transposed for B-operand) ----
__global__ void prepA_kernel(const float* __restrict__ A,
                             unsigned short* __restrict__ AbfT) {
    size_t idx = (size_t)blockIdx.x * 256 + threadIdx.x;   // t*D*N + d*N + n
    int n = (int)(idx & (NPTS - 1));
    size_t rem = idx >> 12;
    int d = (int)(rem & (DDIM - 1));
    int t = (int)(rem >> 8);
    AbfT[idx] = f32_to_bf16_rne(A[((size_t)t * NPTS + (size_t)n) * DDIM + d]);
}

// ---- affine part in exact fp32: W[i,d] = b[d] + sum_k Z[i,k]*Aaff[d,k] ----
__global__ void affine_kernel(const float* __restrict__ Z,
                              const float* __restrict__ Aaff,
                              const float* __restrict__ baff,
                              float* __restrict__ Wout) {
    __shared__ float zrow[DDIM];
    const int i = blockIdx.x;
    const int d = threadIdx.x;
    zrow[d] = Z[(size_t)i * DDIM + d];
    __syncthreads();
    const float* ar = Aaff + (size_t)d * DDIM;
    float s = baff[d];
#pragma unroll 8
    for (int k = 0; k < DDIM; k += 4) {
        float4 a4 = *(const float4*)(ar + k);
        s = fmaf(zrow[k + 0], a4.x, s);
        s = fmaf(zrow[k + 1], a4.y, s);
        s = fmaf(zrow[k + 2], a4.z, s);
        s = fmaf(zrow[k + 3], a4.w, s);
    }
    Wout[(size_t)i * DDIM + d] = s;
}

// ---- fused flow step: S = Zi*Zj^T (WMMA) -> K=exp -> acc += K*A_t (WMMA) ----
#define ZJP 264   // 64-row tile of Zbf, pitch in bf16 elements (16B multiple, conflict-free)
#define AJP 72    // 128-row tile of AbfT
#define KTP 72    // per-wave 16x64 K tile

__launch_bounds__(256)
__global__ void flow_kernel(const float* __restrict__ Zcur,
                            const unsigned short* __restrict__ Zbf,
                            const float* __restrict__ sq,
                            const unsigned short* __restrict__ AbfT, // [DDIM][NPTS] for step t
                            const float* __restrict__ Wbuf,
                            float* __restrict__ Znext) {
    __shared__ __align__(16) unsigned short ZJ[64 * ZJP];
    __shared__ __align__(16) unsigned short AJ[128 * AJP];
    __shared__ __align__(16) unsigned short KT[8 * 16 * KTP];

    const int tid  = threadIdx.x;
    const int wave = tid >> 5;
    const int lane = tid & 31;
    const int ln   = lane & 15;
    const int hf   = lane >> 4;          // lane half (0/1)
    const int i0   = blockIdx.x * 128;   // 128 rows per block (8 waves x 16)
    const int d0   = blockIdx.y * 128;   // d-split of 2
    const int iw   = i0 + wave * 16;

    // A-operand of GEMM1: this wave's 16 Z rows, all 8 K-chunks, held in regs.
    BF16x16 Areg[8];
    {
        const v8bf* zr = (const v8bf*)(Zbf + (size_t)(iw + ln) * DDIM);
#pragma unroll
        for (int kc = 0; kc < 8; ++kc) {
            Areg[kc].h[0] = zr[kc * 4 + hf];        // K = k0 + hf*8 .. +7
            Areg[kc].h[1] = zr[kc * 4 + 2 + hf];    // K = k0+16+hf*8 .. +7
        }
    }
    float sqi[8];
#pragma unroll
    for (int r = 0; r < 8; ++r)
        sqi[r] = sq[iw + r + hf * 8];

    v8f acc[8];
#pragma unroll
    for (int dt = 0; dt < 8; ++dt)
#pragma unroll
        for (int r = 0; r < 8; ++r)
            acc[dt][r] = 0.f;

    unsigned short* KTw = KT + wave * 16 * KTP;

    for (int jb = 0; jb < NPTS / 64; ++jb) {
        const int j0 = jb * 64;
        __syncthreads();
        // cooperative stage: Zbf j-tile (64x256) into ZJ
#pragma unroll
        for (int it = 0; it < 8; ++it) {
            int c  = tid + it * 256;
            int r  = c >> 5;
            int cc = c & 31;
            *(v8bf*)&ZJ[r * ZJP + cc * 8] =
                ((const v8bf*)(Zbf + (size_t)(j0 + r) * DDIM))[cc];
        }
        // cooperative stage: AbfT tile (rows d0..d0+127, cols j0..j0+63) into AJ
#pragma unroll
        for (int it = 0; it < 4; ++it) {
            int c  = tid + it * 256;
            int r  = c >> 3;
            int cc = c & 7;
            *(v8bf*)&AJ[r * AJP + cc * 8] =
                ((const v8bf*)(AbfT + (size_t)(d0 + r) * NPTS + j0))[cc];
        }
        __syncthreads();

        // GEMM1: S(16x64) = Zi(16x256) x Zj^T(256x64), bf16 in / f32 out
        v8f S[4];
#pragma unroll
        for (int nt = 0; nt < 4; ++nt)
#pragma unroll
            for (int r = 0; r < 8; ++r)
                S[nt][r] = 0.f;
#pragma unroll
        for (int kc = 0; kc < 8; ++kc) {
#pragma unroll
            for (int nt = 0; nt < 4; ++nt) {
                const unsigned short* bp =
                    &ZJ[(nt * 16 + ln) * ZJP + kc * 32 + hf * 8];
                BF16x16 B;
                B.h[0] = *(const v8bf*)bp;
                B.h[1] = *(const v8bf*)(bp + 16);
                S[nt] = __builtin_amdgcn_wmma_f32_16x16x32_bf16(
                    false, Areg[kc].v, false, B.v, (short)0, S[nt], false, false);
            }
        }

        // K = exp((2*dot - sqi - sqj)/512), write bf16 tile to LDS (row-major 16x64)
#pragma unroll
        for (int nt = 0; nt < 4; ++nt) {
            float sqj = sq[j0 + nt * 16 + ln];
#pragma unroll
            for (int r = 0; r < 8; ++r) {
                float e = __expf((2.f * S[nt][r] - sqi[r] - sqj) * (1.f / 512.f));
                KTw[(r + hf * 8) * KTP + nt * 16 + ln] = f32_to_bf16_rne(e);
            }
        }

        // GEMM2: acc(16x128) += K(16x64) x A_t(64x128)
#pragma unroll
        for (int kc2 = 0; kc2 < 2; ++kc2) {
            const unsigned short* kp = &KTw[ln * KTP + kc2 * 32 + hf * 8];
            BF16x16 AK;
            AK.h[0] = *(const v8bf*)kp;          // K = k0 + hf*8 .. +7
            AK.h[1] = *(const v8bf*)(kp + 16);   // K = k0+16+hf*8 .. +7
#pragma unroll
            for (int dt = 0; dt < 8; ++dt) {
                const unsigned short* bp =
                    &AJ[(dt * 16 + ln) * AJP + kc2 * 32 + hf * 8];
                BF16x16 B;
                B.h[0] = *(const v8bf*)bp;
                B.h[1] = *(const v8bf*)(bp + 16);
                acc[dt] = __builtin_amdgcn_wmma_f32_16x16x32_bf16(
                    false, AK.v, false, B.v, (short)0, acc[dt], false, false);
            }
        }
    }

    // epilogue: Znext = Zcur + DT*(W + K@A)
#pragma unroll
    for (int dt = 0; dt < 8; ++dt) {
#pragma unroll
        for (int r = 0; r < 8; ++r) {
            int row = iw + r + hf * 8;
            int col = d0 + dt * 16 + ln;
            size_t idx = (size_t)row * DDIM + col;
            Znext[idx] = Zcur[idx] + 0.125f * (Wbuf[idx] + acc[dt][r]);
        }
    }
}

extern "C" void kernel_launch(void* const* d_in, const int* in_sizes, int n_in,
                              void* d_out, int out_size, void* d_ws, size_t ws_size,
                              hipStream_t stream) {
    const float* X    = (const float*)d_in[0];
    const float* A    = (const float*)d_in[1];
    const float* Aaff = (const float*)d_in[2];
    const float* baff = (const float*)d_in[3];
    float* out = (float*)d_out;
    char*  ws  = (char*)d_ws;

    // ws layout: [0,4M) Z ping buffer | [4M,8M) W | [8M,+16K) sq |
    //            [9M,11M) Zbf (bf16)  | [12M,28M) AbfT (bf16, T*D*N)
    float*          BufA = (float*)(ws);
    float*          Wbuf = (float*)(ws + ((size_t)4 << 20));
    float*          sqp  = (float*)(ws + ((size_t)8 << 20));
    unsigned short* Zbf  = (unsigned short*)(ws + ((size_t)9 << 20));
    unsigned short* AbfT = (unsigned short*)(ws + ((size_t)12 << 20));

    prepA_kernel<<<(TSTEPS * NPTS * DDIM) / 256, 256, 0, stream>>>(A, AbfT);

    const float* Zc = X;
    for (int t = 0; t < TSTEPS; ++t) {
        float* Zn = (t & 1) ? out : BufA;   // t=7 (last) lands in d_out
        prepz_kernel<<<NPTS / 8, 256, 0, stream>>>(Zc, Zbf, sqp);
        affine_kernel<<<NPTS, 256, 0, stream>>>(
            Zc, Aaff + (size_t)t * DDIM * DDIM, baff + (size_t)t * DDIM, Wbuf);
        dim3 grid(NPTS / 128, 2);
        flow_kernel<<<grid, 256, 0, stream>>>(
            Zc, Zbf, sqp, AbfT + (size_t)t * DDIM * NPTS, Wbuf, Zn);
        Zc = Zn;
    }
}
